// Model_26654567039432
// MI455X (gfx1250) — matrix operand order repaired and checked by reading, not compiled, not run on hardware
//
#include <hip/hip_runtime.h>
#include <math.h>

// ---------------------------------------------------------------------------
// Point-cloud segmentation network (diff-attention U-Net) for gfx1250.
// All dense contractions go through one WMMA (v_wmma_f32_16x16x32_f16) GEMM
// with LDS-staged, double-buffered tiles copied by gfx1250 async
// global->LDS instructions (ASYNCcnt / s_wait_asynccnt).
// Wave32: one wave computes one 16x16 output tile; 4 waves per block.
// ---------------------------------------------------------------------------

#define USE_ASYNC_COPY 1   // flip to 0 if the assembler rejects the async ops

#define BN   4
#define NPTS 2048
#define CAT  16
#define NCLS 50
#define SA_K 20
#define SX   ((long)NPTS * 3)   // batch stride of xyz (all x-levels are views)

// LDS tile strides (floats). Multiples of 4 floats -> 16B-aligned rows for
// async b128 destinations and ds_load_b128 fragment reads; chosen so rows
// land on distinct banks across lanes (36*r mod 64 and 68*r mod 64 spread).
#define ASTR  36   // A tile [row(16)][k(32)]
#define B0STR 68   // B tile btrans=0: [k(32)][n(64)]
#define B1STR 36   // B tile btrans=1: [n(64)][k(32)]

typedef __attribute__((ext_vector_type(16))) _Float16 v16h;
typedef __attribute__((ext_vector_type(8)))  float    v8f;

__device__ __forceinline__ float gelu_f(float x) {
  const float k0 = 0.7978845608028654f; // sqrt(2/pi), tanh-approx gelu (jax default)
  return 0.5f * x * (1.f + tanhf(k0 * (x + 0.044715f * x * x * x)));
}

// ---- generic WMMA GEMM: C[b] = act(scale * A[b] @ B[b](^T) + bias) --------
// A: (M,K) row-major, row stride lda, batch stride sA
// B: (K,Ncol) ldb=Ncol (btrans=0) or row-major (Ncol,K) ldb=K (btrans=1)
// Block = 4 waves = 16 rows x 64 cols of C. K consumed in chunks of 32.
// Tiles staged in LDS (zero-filled on edges) -> branch-free WMMA fragments.
__global__ void __launch_bounds__(128)
k_gemm(const float* __restrict__ A, const float* __restrict__ Bm,
       const float* __restrict__ bias, float* __restrict__ C,
       int M, int K, int Ncol, int lda, int ldb, int ldc,
       long sA, long sB, long sC, int btrans, int act, float scale)
{
  __shared__ float Ash[2][16 * ASTR];    // 576 floats per buffer
  __shared__ float Bsh[2][64 * B1STR];   // 2304 floats >= 32*B0STR = 2176

  const int tid  = threadIdx.y * 32 + threadIdx.x;
  const int lane = threadIdx.x & 31;
  const int r0   = blockIdx.x << 4;
  const int cBlk = blockIdx.y << 6;

  A  += (long)blockIdx.z * sA;
  Bm += (long)blockIdx.z * sB;
  C  += (long)blockIdx.z * sC;

  const int  nCh    = (K + 31) >> 5;
  const bool rowsOK = (r0 + 16) <= M;
  const bool colsOK = (cBlk + 64) <= Ncol;
  const bool ldaOK  = (lda & 3) == 0;
  const bool ldbOK  = (ldb & 3) == 0;

  auto isClean = [&](int ch) {
    return ((ch << 5) + 32 <= K) && rowsOK && colsOK && ldaOK && ldbOK;
  };

  // Stage chunk ch into LDS buffer ch&1. Uniform control flow across block.
  auto stage = [&](int ch) {
    const int kb = ch << 5;
    float* ash = Ash[ch & 1];
    float* bsh = Bsh[ch & 1];
    if (isClean(ch)) {
#if USE_ASYNC_COPY
      { // A tile: 128 float4 segments, one per thread (async copy to LDS)
        int r = tid >> 3, k4 = (tid & 7) << 2;
        const float* gp = A + (long)(r0 + r) * lda + kb + k4;
        unsigned lo = (unsigned)(size_t)&ash[r * ASTR + k4];
        asm volatile("global_load_async_to_lds_b128 %0, %1, off"
                     :: "v"(lo), "v"(gp) : "memory");
      }
      // B tile: 512 float4 segments, 4 per thread; branch on uniform btrans
      if (btrans) {
#pragma unroll
        for (int s = 0; s < 4; ++s) {
          int q = tid + (s << 7);
          int c = q >> 3, r4 = (q & 7) << 2;
          const float* gp = Bm + (long)(cBlk + c) * ldb + kb + r4;
          unsigned lo = (unsigned)(size_t)&bsh[c * B1STR + r4];
          asm volatile("global_load_async_to_lds_b128 %0, %1, off"
                       :: "v"(lo), "v"(gp) : "memory");
        }
      } else {
#pragma unroll
        for (int s = 0; s < 4; ++s) {
          int q = tid + (s << 7);
          int r = q >> 4, c4 = (q & 15) << 2;
          const float* gp = Bm + (long)(kb + r) * ldb + cBlk + c4;
          unsigned lo = (unsigned)(size_t)&bsh[r * B0STR + c4];
          asm volatile("global_load_async_to_lds_b128 %0, %1, off"
                       :: "v"(lo), "v"(gp) : "memory");
        }
      }
#else
#pragma unroll
      for (int s = 0; s < 4; ++s) { // A tile, plain loads
        int e = tid + (s << 7);
        int r = e >> 5, k = e & 31;
        ash[r * ASTR + k] = A[(long)(r0 + r) * lda + kb + k];
      }
      if (btrans) {
#pragma unroll
        for (int s = 0; s < 16; ++s) {
          int e = tid + (s << 7);
          int r = e >> 6, c = e & 63;
          bsh[c * B1STR + r] = Bm[(long)(cBlk + c) * ldb + kb + r];
        }
      } else {
#pragma unroll
        for (int s = 0; s < 16; ++s) {
          int e = tid + (s << 7);
          int r = e >> 6, c = e & 63;
          bsh[r * B0STR + c] = Bm[(long)(kb + r) * ldb + cBlk + c];
        }
      }
#endif
    } else {
      // edge path: clamped addresses + select (no divergent load predication)
#pragma unroll
      for (int s = 0; s < 4; ++s) { // A: 512 elements
        int e = tid + (s << 7);
        int r = e >> 5, k = e & 31;
        int gr = r0 + r, gk = kb + k;
        bool ok = (gr < M) && (gk < K);
        float v = A[(long)(ok ? gr : 0) * lda + (ok ? gk : 0)];
        ash[r * ASTR + k] = ok ? v : 0.f;
      }
      if (btrans) {
#pragma unroll
        for (int s = 0; s < 16; ++s) { // B: 2048 elements
          int e = tid + (s << 7);
          int r = e >> 6, c = e & 63;        // r = k index, c = n index
          int gk = kb + r, gc = cBlk + c;
          bool ok = (gk < K) && (gc < Ncol);
          float v = Bm[(long)(ok ? gc : 0) * ldb + (ok ? gk : 0)];
          bsh[c * B1STR + r] = ok ? v : 0.f;
        }
      } else {
#pragma unroll
        for (int s = 0; s < 16; ++s) {
          int e = tid + (s << 7);
          int r = e >> 6, c = e & 63;
          int gk = kb + r, gc = cBlk + c;
          bool ok = (gk < K) && (gc < Ncol);
          float v = Bm[(long)(ok ? gk : 0) * ldb + (ok ? gc : 0)];
          bsh[r * B0STR + c] = ok ? v : 0.f;
        }
      }
    }
  };

  const int cl   = (threadIdx.y << 4) + (lane & 15);  // col within 64-wide stripe
  const int arow = lane & 15;
  const int hiA  = (lane >> 4) << 4;   // lanes 16-31 hold K+16 block (A frag)
  const int hiB  = (lane >> 4) << 4;   // lanes 16-31 hold K+16 block (B frag)

  stage(0);
  v8f acc = {};
  for (int ch = 0; ch < nCh; ++ch) {
    bool haveNext = (ch + 1) < nCh;
    if (haveNext) stage(ch + 1);
#if USE_ASYNC_COPY
    // wait for chunk ch's async copies (in-order); next chunk may keep 5 in flight
    if (haveNext && isClean(ch + 1))
      asm volatile("s_wait_asynccnt 0x5" ::: "memory");
    else
      asm volatile("s_wait_asynccnt 0x0" ::: "memory");
#endif
    __syncthreads();
    if (ch + 2 < nCh && rowsOK)
      __builtin_prefetch(A + (long)(r0 + arow) * lda + ((ch + 2) << 5), 0, 1);

    const float* ash = Ash[ch & 1];
    const float* bsh = Bsh[ch & 1];

    // A fragment: 4x ds_load_b128 (rows 16B aligned), then convert
    const float* ar = &ash[arow * ASTR + hiA];
    float4 a0 = *(const float4*)(ar);
    float4 a1 = *(const float4*)(ar + 4);
    float4 a2 = *(const float4*)(ar + 8);
    float4 a3 = *(const float4*)(ar + 12);
    v16h a;
    a[0]  = (_Float16)a0.x; a[1]  = (_Float16)a0.y;
    a[2]  = (_Float16)a0.z; a[3]  = (_Float16)a0.w;
    a[4]  = (_Float16)a1.x; a[5]  = (_Float16)a1.y;
    a[6]  = (_Float16)a1.z; a[7]  = (_Float16)a1.w;
    a[8]  = (_Float16)a2.x; a[9]  = (_Float16)a2.y;
    a[10] = (_Float16)a2.z; a[11] = (_Float16)a2.w;
    a[12] = (_Float16)a3.x; a[13] = (_Float16)a3.y;
    a[14] = (_Float16)a3.z; a[15] = (_Float16)a3.w;

    v16h b;
    if (btrans) {
      // contiguous: 4x ds_load_b128 then convert
      const float* br = &bsh[cl * B1STR + hiB];
      float4 b0 = *(const float4*)(br);
      float4 b1 = *(const float4*)(br + 4);
      float4 b2 = *(const float4*)(br + 8);
      float4 b3 = *(const float4*)(br + 12);
      b[0]  = (_Float16)b0.x; b[1]  = (_Float16)b0.y;
      b[2]  = (_Float16)b0.z; b[3]  = (_Float16)b0.w;
      b[4]  = (_Float16)b1.x; b[5]  = (_Float16)b1.y;
      b[6]  = (_Float16)b1.z; b[7]  = (_Float16)b1.w;
      b[8]  = (_Float16)b2.x; b[9]  = (_Float16)b2.y;
      b[10] = (_Float16)b2.z; b[11] = (_Float16)b2.w;
      b[12] = (_Float16)b3.x; b[13] = (_Float16)b3.y;
      b[14] = (_Float16)b3.z; b[15] = (_Float16)b3.w;
    } else {
      // strided column reads: batch 16 loads into registers, then convert
      float tmp[16];
#pragma unroll
      for (int j = 0; j < 16; ++j) tmp[j] = bsh[(hiB + j) * B0STR + cl];
#pragma unroll
      for (int j = 0; j < 16; ++j) b[j] = (_Float16)tmp[j];
    }

    acc = __builtin_amdgcn_wmma_f32_16x16x32_f16(false, a, false, b,
                                                 (short)0, acc, false, false);
    __syncthreads();   // buffer ch&1 free for reuse at chunk ch+2
  }

#pragma unroll
  for (int r = 0; r < 8; ++r) {              // C: VGPR r -> rows r / r+8
    int rr = r0 + r + ((lane >> 4) << 3);
    int cc = cBlk + cl;
    if (rr < M && cc < Ncol) {
      float v = acc[r] * scale;
      if (bias) v += bias[cc];
      if      (act == 1) v = gelu_f(v);
      else if (act == 2) v = fmaxf(v, 0.f);
      else if (act == 3) v = 1.f / (1.f + __expf(-v));
      C[(long)rr * ldc + cc] = v;
    }
  }
}

// ---- SA module: 20-NN with radius filter ----------------------------------
__global__ void k_sa_knn(const float* __restrict__ xyz, int* __restrict__ idx,
                         float r2)
{
  int t = blockIdx.x * blockDim.x + threadIdx.x;
  if (t >= BN * NPTS) return;
  int b = t / NPTS, m = t - b * NPTS;
  const float* xb = xyz + (long)b * SX;
  float mx = xb[m*3+0], my = xb[m*3+1], mz = xb[m*3+2];
  float bd[SA_K]; int bi[SA_K];
#pragma unroll
  for (int k = 0; k < SA_K; ++k) { bd[k] = 1e30f; bi[k] = 0; }
  for (int n = 0; n < NPTS; ++n) {
    float dx = xb[n*3+0]-mx, dy = xb[n*3+1]-my, dz = xb[n*3+2]-mz;
    float d = dx*dx + dy*dy + dz*dz;
    if (d < bd[SA_K-1]) {
      bd[SA_K-1] = d; bi[SA_K-1] = n;
#pragma unroll
      for (int j = SA_K-1; j > 0; --j) {
        if (bd[j] < bd[j-1]) {
          float td = bd[j]; bd[j] = bd[j-1]; bd[j-1] = td;
          int   ti = bi[j]; bi[j] = bi[j-1]; bi[j-1] = ti;
        }
      }
    }
  }
  int i0 = bi[0];
#pragma unroll
  for (int k = 0; k < SA_K; ++k)
    idx[(long)t * SA_K + k] = (bd[k] <= r2) ? bi[k] : i0;
}

__global__ void k_sa_group(const float* __restrict__ xyz,
                           const float* __restrict__ f,
                           const int* __restrict__ idx, float* __restrict__ g)
{
  long t = (long)blockIdx.x * blockDim.x + threadIdx.x;
  if (t >= (long)BN * NPTS * SA_K) return;
  long bm = t / SA_K;
  int b = (int)(bm / NPTS), m = (int)(bm - (long)b * NPTS);
  int nn = idx[t];
  const float* xb = xyz + (long)b * SX;
  float* go = g + t * 35;
  go[0] = xb[nn*3+0] - xb[m*3+0];
  go[1] = xb[nn*3+1] - xb[m*3+1];
  go[2] = xb[nn*3+2] - xb[m*3+2];
  const float* fp = f + ((long)b * NPTS + nn) * 32;
#pragma unroll
  for (int c = 0; c < 32; ++c) go[3 + c] = fp[c];
}

__global__ void k_maxk(const float* __restrict__ g, float* __restrict__ out,
                       long total)
{
  long t = (long)blockIdx.x * blockDim.x + threadIdx.x;
  if (t >= total) return;
  int c = (int)(t & 31);
  long bm = t >> 5;
  const float* gp = g + bm * (SA_K * 32) + c;
  float v = -3.0e38f;
#pragma unroll
  for (int k = 0; k < SA_K; ++k) v = fmaxf(v, gp[k * 32]);
  out[t] = v;
}

// ---- radius-masked softmax (in place over attention logits) ---------------
__global__ void k_softmax_mask(float* __restrict__ logits,
                               const float* __restrict__ xyz, long sXyz,
                               int M, int Nv, float r2)
{
  int bm = blockIdx.x;
  int b = bm / M, m = bm - b * M;
  float* row = logits + (long)bm * Nv;
  const float* xb = xyz + (long)b * sXyz;
  float cx = xb[m*3+0], cy = xb[m*3+1], cz = xb[m*3+2];
  __shared__ float red[256];
  int tid = threadIdx.x;
  float lmax = -3.0e38f;
  for (int n = tid; n < Nv; n += blockDim.x) {
    float dx = xb[n*3+0]-cx, dy = xb[n*3+1]-cy, dz = xb[n*3+2]-cz;
    if (dx*dx + dy*dy + dz*dz <= r2) lmax = fmaxf(lmax, row[n]);
  }
  red[tid] = lmax; __syncthreads();
  for (int s = blockDim.x >> 1; s > 0; s >>= 1) {
    if (tid < s) red[tid] = fmaxf(red[tid], red[tid + s]);
    __syncthreads();
  }
  float rmax = red[0]; __syncthreads();
  float lsum = 0.f;
  for (int n = tid; n < Nv; n += blockDim.x) {
    float dx = xb[n*3+0]-cx, dy = xb[n*3+1]-cy, dz = xb[n*3+2]-cz;
    float v = (dx*dx + dy*dy + dz*dz <= r2) ? __expf(row[n] - rmax) : 0.f;
    row[n] = v; lsum += v;
  }
  red[tid] = lsum; __syncthreads();
  for (int s = blockDim.x >> 1; s > 0; s >>= 1) {
    if (tid < s) red[tid] += red[tid + s];
    __syncthreads();
  }
  float inv = 1.f / red[0];
  for (int n = tid; n < Nv; n += blockDim.x) row[n] *= inv;
}

// ---- concat / gather / pooling helpers ------------------------------------
__global__ void k_concat2(const float* __restrict__ A, long sA, int C1,
                          const float* __restrict__ Bb, long sB, int C2,
                          float* __restrict__ O, int M, long total)
{
  long t = (long)blockIdx.x * blockDim.x + threadIdx.x;
  if (t >= total) return;
  int Ct = C1 + C2;
  int c = (int)(t % Ct); long bm = t / Ct;
  int m = (int)(bm % M); int b = (int)(bm / M);
  O[t] = (c < C1) ? A[(long)b*sA + (long)m*C1 + c]
                  : Bb[(long)b*sB + (long)m*C2 + (c - C1)];
}

__global__ void k_concat3(const float* __restrict__ A, long sA, int C1,
                          const float* __restrict__ Bb, long sB, int C2,
                          const float* __restrict__ Cc, long sC, int C3,
                          float* __restrict__ O, int M, long total)
{
  long t = (long)blockIdx.x * blockDim.x + threadIdx.x;
  if (t >= total) return;
  int Ct = C1 + C2 + C3;
  int c = (int)(t % Ct); long bm = t / Ct;
  int m = (int)(bm % M); int b = (int)(bm / M);
  float v;
  if      (c < C1)      v = A [(long)b*sA + (long)m*C1 + c];
  else if (c < C1 + C2) v = Bb[(long)b*sB + (long)m*C2 + (c - C1)];
  else                  v = Cc[(long)b*sC + (c - C1 - C2)];   // broadcast row
  O[t] = v;
}

__global__ void k_rowmax(const float* __restrict__ A, long sA, int M, int C,
                         float* __restrict__ out, int ldo, int off)
{
  int t = blockIdx.x * blockDim.x + threadIdx.x;
  if (t >= BN * C) return;
  int b = t / C, c = t - b * C;
  const float* ab = A + (long)b * sA + c;
  float v = -3.0e38f;
  for (int m = 0; m < M; ++m) v = fmaxf(v, ab[(long)m * C]);
  out[(long)b * ldo + off + c] = v;
}

__global__ void k_copyvec(const float* __restrict__ in, int Ci,
                          float* __restrict__ out, int ldo, int off, int total)
{
  int t = blockIdx.x * blockDim.x + threadIdx.x;
  if (t >= total) return;
  int b = t / Ci, c = t - b * Ci;
  out[(long)b * ldo + off + c] = in[t];
}

// ---- FP (feature propagation) helpers -------------------------------------
__global__ void k_knn3(const float* __restrict__ xyz1, long s1,
                       const float* __restrict__ xyz2, long s2,
                       int M1, int M2, int* __restrict__ idx3,
                       float* __restrict__ w3)
{
  int t = blockIdx.x * blockDim.x + threadIdx.x;
  if (t >= BN * M1) return;
  int b = t / M1, m = t - b * M1;
  const float* p = xyz1 + (long)b * s1 + (long)m * 3;
  float px = p[0], py = p[1], pz = p[2];
  const float* xb = xyz2 + (long)b * s2;
  float bd[3] = {1e30f, 1e30f, 1e30f}; int bi[3] = {0, 0, 0};
  for (int n = 0; n < M2; ++n) {
    float dx = xb[n*3+0]-px, dy = xb[n*3+1]-py, dz = xb[n*3+2]-pz;
    float d = dx*dx + dy*dy + dz*dz;
    if (d < bd[2]) {
      bd[2] = d; bi[2] = n;
      if (bd[2] < bd[1]) { float td=bd[2];bd[2]=bd[1];bd[1]=td; int ti=bi[2];bi[2]=bi[1];bi[1]=ti; }
      if (bd[1] < bd[0]) { float td=bd[1];bd[1]=bd[0];bd[0]=td; int ti=bi[1];bi[1]=bi[0];bi[0]=ti; }
    }
  }
  float w0 = 1.f/(bd[0]+1e-8f), w1 = 1.f/(bd[1]+1e-8f), w2 = 1.f/(bd[2]+1e-8f);
  float ws = w0 + w1 + w2;
  idx3[t*3+0] = bi[0]; idx3[t*3+1] = bi[1]; idx3[t*3+2] = bi[2];
  w3[t*3+0] = w0/ws;   w3[t*3+1] = w1/ws;   w3[t*3+2] = w2/ws;
}

__global__ void k_interp_concat(const float* __restrict__ f1, long sf1, int C1,
                                const float* __restrict__ f2, long sf2, int C2,
                                const int* __restrict__ idx3,
                                const float* __restrict__ w3,
                                float* __restrict__ h, int M1, long total)
{
  long t = (long)blockIdx.x * blockDim.x + threadIdx.x;
  if (t >= total) return;
  int Ct = C1 + C2;
  int c = (int)(t % Ct); long bm = t / Ct;
  int m = (int)(bm % M1); int b = (int)(bm / M1);
  float v;
  if (c < C2) {                               // interpolated feat2 first
    const int*   id = idx3 + bm * 3;
    const float* w  = w3   + bm * 3;
    const float* fb = f2 + (long)b * sf2;
    v = w[0]*fb[(long)id[0]*C2 + c] + w[1]*fb[(long)id[1]*C2 + c]
      + w[2]*fb[(long)id[2]*C2 + c];
  } else {
    v = f1[(long)b*sf1 + (long)m*C1 + (c - C2)];
  }
  h[t] = v;
}

// ---- head helpers ---------------------------------------------------------
__global__ void k_mean_rows(const float* __restrict__ A, int M, int C,
                            float* __restrict__ out)
{
  int t = blockIdx.x * blockDim.x + threadIdx.x;
  if (t >= BN * C) return;
  int b = t / C, c = t - b * C;
  const float* ab = A + (long)b * M * C + c;
  float s = 0.f;
  for (int m = 0; m < M; ++m) s += ab[(long)m * C];
  out[t] = s / (float)M;
}

__global__ void k_rowscale(const float* __restrict__ A,
                           const float* __restrict__ s,
                           float* __restrict__ O, int M, int C, long total)
{
  long t = (long)blockIdx.x * blockDim.x + threadIdx.x;
  if (t >= total) return;
  int c = (int)(t % C); long bm = t / C; int b = (int)(bm / M);
  O[t] = A[t] * s[(long)b * C + c];
}

__global__ void k_transpose_out(const float* __restrict__ A,
                                float* __restrict__ O, int M, int C)
{
  long t = (long)blockIdx.x * blockDim.x + threadIdx.x;
  if (t >= (long)BN * M * C) return;
  int c = (int)(t % C); long bm = t / C;
  int m = (int)(bm % M); int b = (int)(bm / M);
  O[(long)b * C * M + (long)c * M + m] = A[t];
}

// ---------------------------------------------------------------------------
// Host-side orchestration
// ---------------------------------------------------------------------------
static inline unsigned cdiv(long a, long b) { return (unsigned)((a + b - 1) / b); }

static void gemm(hipStream_t st, int nb,
                 const float* A, long sA, int lda,
                 const float* Bm, long sB, int ldb, int btrans,
                 const float* bias, float* C, long sC, int ldc,
                 int M, int K, int Ncol, int act, float scale)
{
  dim3 g(cdiv(M, 16), cdiv(Ncol, 64), nb);
  dim3 b(32, 4, 1);
  k_gemm<<<g, b, 0, st>>>(A, Bm, bias, C, M, K, Ncol, lda, ldb, ldc,
                          sA, sB, sC, btrans, act, scale);
}

static float* balloc(char*& p, size_t nflt) {
  float* r = (float*)p;
  p += ((nflt * sizeof(float) + 255) & ~(size_t)255);
  return r;
}

// diff_conv: masked-attention conv. feat has Nv rows/batch (stride sFeat);
// query set = first Mv rows. xyz is the shared coordinate base (views).
static void diff_conv(hipStream_t st, const float* feat, long sFeat, int Nv,
                      const float* xyz, int Mv, int C, int Cout,
                      const float* wq, const float* wk,
                      const float* ow, const float* ob,
                      float r2, float* out, long sOut, char* scr)
{
  float* qb = balloc(scr, (size_t)BN * Mv * C);
  float* kb = balloc(scr, (size_t)BN * Nv * C);
  float* lg = balloc(scr, (size_t)BN * Mv * Nv);
  float* ag = balloc(scr, (size_t)BN * Mv * C);
  float* hb = balloc(scr, (size_t)BN * Mv * 2 * C);
  float inv = 1.f / sqrtf((float)C);
  gemm(st, BN, feat, sFeat, C, wq, 0, C, 0, nullptr, qb, (long)Mv*C, C, Mv, C, C, 0, 1.f);
  gemm(st, BN, feat, sFeat, C, wk, 0, C, 0, nullptr, kb, (long)Nv*C, C, Nv, C, C, 0, 1.f);
  gemm(st, BN, qb, (long)Mv*C, C, kb, (long)Nv*C, C, 1, nullptr,
       lg, (long)Mv*Nv, Nv, Mv, C, Nv, 0, inv);                 // q @ k^T
  k_softmax_mask<<<dim3(BN * Mv), dim3(256), 0, st>>>(lg, xyz, SX, Mv, Nv, r2);
  gemm(st, BN, lg, (long)Mv*Nv, Nv, feat, sFeat, C, 0, nullptr,
       ag, (long)Mv*C, C, Mv, Nv, C, 0, 1.f);                   // a @ feat
  long tot = (long)BN * Mv * 2 * C;
  k_concat2<<<cdiv(tot, 256), 256, 0, st>>>(ag, (long)Mv*C, C,
                                            feat, sFeat, C, hb, Mv, tot);
  gemm(st, BN, hb, (long)Mv*2*C, 2*C, ow, 0, Cout, 0, ob,
       out, sOut, Cout, Mv, 2*C, Cout, 1, 1.f);                 // gelu
}

static void fp_mod(hipStream_t st, const float* xyz, int M1, int M2,
                   const float* f1, long sf1, int C1,
                   const float* f2, long sf2, int C2,
                   const float* w, const float* bias, int Cout,
                   float* out, long sOut, char* scr)
{
  int*   idx3 = (int*)balloc(scr, (size_t)BN * M1 * 3);
  float* w3   = balloc(scr, (size_t)BN * M1 * 3);
  float* hb   = balloc(scr, (size_t)BN * M1 * (C1 + C2));
  k_knn3<<<cdiv(BN * M1, 256), 256, 0, st>>>(xyz, SX, xyz, SX, M1, M2, idx3, w3);
  long tot = (long)BN * M1 * (C1 + C2);
  k_interp_concat<<<cdiv(tot, 256), 256, 0, st>>>(f1, sf1, C1, f2, sf2, C2,
                                                  idx3, w3, hb, M1, tot);
  gemm(st, BN, hb, (long)M1*(C1+C2), C1+C2, w, 0, Cout, 0, bias,
       out, sOut, Cout, M1, C1+C2, Cout, 1, 1.f);
}

// param indices: depth-first insertion order of setup_inputs()
enum {
  IN_X = 0, IN_L,
  LE0_W, LE0_B, LE1A_W, LE1A_B, LE1B_W, LE1B_B,
  C1_WQ, C1_WK, C1_OW, C1_OB,  C2_WQ, C2_WK, C2_OW, C2_OB,
  C3_WQ, C3_WK, C3_OW, C3_OB,  C4_WQ, C4_WK, C4_OW, C4_OB,
  C5_WQ, C5_WK, C5_OW, C5_OB,
  FP4_W, FP4_B, UP5_WQ, UP5_WK, UP5_OW, UP5_OB,
  FP3_W, FP3_B, UP4_WQ, UP4_WK, UP4_OW, UP4_OB,
  FP2_W, FP2_B, UP3_WQ, UP3_WK, UP3_OW, UP3_OB,
  FP1_W, FP1_B, UP2_WQ, UP2_WK, UP2_OW, UP2_OB,
  G1A_W, G1A_B, G1B_W, G1B_B, G2A_W, G2A_B, G2B_W, G2B_B,
  UP1_W, UP1_B, SE1, SE2, LAST, N_IN
};

extern "C" void kernel_launch(void* const* d_in, const int* in_sizes, int n_in,
                              void* d_out, int out_size, void* d_ws, size_t ws_size,
                              hipStream_t stream)
{
  (void)in_sizes; (void)n_in; (void)out_size; (void)ws_size;
#define P(n) ((const float*)d_in[(n)])
  const float* X = P(IN_X);
  const float* L = P(IN_L);

  // persistent buffers (~50 MB), transient scratch overlaid after them.
  char* wp = (char*)d_ws;
  float* f0     = balloc(wp, (size_t)BN*NPTS*32);
  float* f1s    = balloc(wp, (size_t)BN*NPTS*32);
  float* f1     = balloc(wp, (size_t)BN*NPTS*64);
  float* f2     = balloc(wp, (size_t)BN*512*128);
  float* f3     = balloc(wp, (size_t)BN*256*256);
  float* f4     = balloc(wp, (size_t)BN*128*512);
  float* f5     = balloc(wp, (size_t)BN*64*1024);
  float* fp4o   = balloc(wp, (size_t)BN*128*256);
  float* up5o   = balloc(wp, (size_t)BN*128*512);
  float* fp3o   = balloc(wp, (size_t)BN*256*128);
  float* up4o   = balloc(wp, (size_t)BN*256*256);
  float* fp2o   = balloc(wp, (size_t)BN*512*64);
  float* up3o   = balloc(wp, (size_t)BN*512*128);
  float* fp1o   = balloc(wp, (size_t)BN*NPTS*32);
  float* up2o   = balloc(wp, (size_t)BN*NPTS*64);
  float* emb    = balloc(wp, (size_t)BN*144);
  float* featc  = balloc(wp, (size_t)BN*NPTS*211);
  float* featg  = balloc(wp, (size_t)BN*NPTS*256);
  float* feats  = balloc(wp, (size_t)BN*NPTS*256);
  float* s0     = balloc(wp, (size_t)BN*256);
  float* s1g    = balloc(wp, (size_t)BN*32);
  float* s2s    = balloc(wp, (size_t)BN*256);
  float* lgt    = balloc(wp, (size_t)BN*NPTS*NCLS);
  char* scr = wp;     // transient region reused by every stage (~70 MB peak)

  // ---- le0: f = gelu(x @ W + b) ----
  gemm(stream, BN, X, SX, 3, P(LE0_W), 0, 32, 0, P(LE0_B),
       f0, (long)NPTS*32, 32, NPTS, 3, 32, 1, 1.f);

  // ---- SA module ----
  {
    char* sp = scr;
    int*   idxsa = (int*)balloc(sp, (size_t)BN*NPTS*SA_K);
    float* g0 = balloc(sp, (size_t)BN*NPTS*SA_K*35);
    float* g1 = balloc(sp, (size_t)BN*NPTS*SA_K*32);
    float* g2 = balloc(sp, (size_t)BN*NPTS*SA_K*32);
    k_sa_knn<<<cdiv(BN*NPTS, 256), 256, 0, stream>>>(X, idxsa, 0.0025f);
    long tg = (long)BN*NPTS*SA_K;
    k_sa_group<<<cdiv(tg, 256), 256, 0, stream>>>(X, f0, idxsa, g0);
    int rows = BN*NPTS*SA_K;
    gemm(stream, 1, g0, 0, 35, P(LE1A_W), 0, 32, 0, P(LE1A_B),
         g1, 0, 32, rows, 35, 32, 2, 1.f);                     // relu
    gemm(stream, 1, g1, 0, 32, P(LE1B_W), 0, 32, 0, P(LE1B_B),
         g2, 0, 32, rows, 32, 32, 2, 1.f);
    k_maxk<<<cdiv((long)BN*NPTS*32, 256), 256, 0, stream>>>(g2, f1s,
                                                            (long)BN*NPTS*32);
  }

  // ---- encoder: 5 diff-attention convs ----
  diff_conv(stream, f1s, (long)NPTS*32, NPTS, X, NPTS, 32,  64,
            P(C1_WQ), P(C1_WK), P(C1_OW), P(C1_OB), 0.05f, f1, (long)NPTS*64, scr);
  diff_conv(stream, f1, (long)NPTS*64, NPTS, X, 512, 64, 128,
            P(C2_WQ), P(C2_WK), P(C2_OW), P(C2_OB), 0.20f, f2, 512L*128, scr);
  diff_conv(stream, f2, 512L*128, 512, X, 256, 128, 256,
            P(C3_WQ), P(C3_WK), P(C3_OW), P(C3_OB), 0.40f, f3, 256L*256, scr);
  diff_conv(stream, f3, 256L*256, 256, X, 128, 256, 512,
            P(C4_WQ), P(C4_WK), P(C4_OW), P(C4_OB), 0.80f, f4, 128L*512, scr);
  diff_conv(stream, f4, 128L*512, 128, X, 64, 512, 1024,
            P(C5_WQ), P(C5_WK), P(C5_OW), P(C5_OB), 1.60f, f5, 64L*1024, scr);

  // ---- global embedding (e1, e2, label) ----
  {
    char* sp = scr;
    float* ce1 = balloc(sp, (size_t)BN*64*1027);
    float* t1  = balloc(sp, (size_t)BN*64*256);
    float* t2  = balloc(sp, (size_t)BN*64*64);
    float* ce2 = balloc(sp, (size_t)BN*128*515);
    float* t3  = balloc(sp, (size_t)BN*128*128);
    float* t4  = balloc(sp, (size_t)BN*128*64);
    long tot = (long)BN*64*1027;
    k_concat2<<<cdiv(tot,256),256,0,stream>>>(X, SX, 3, f5, 64L*1024, 1024,
                                              ce1, 64, tot);
    gemm(stream, BN, ce1, 64L*1027, 1027, P(G1A_W), 0, 256, 0, P(G1A_B),
         t1, 64L*256, 256, 64, 1027, 256, 1, 1.f);
    gemm(stream, BN, t1, 64L*256, 256, P(G1B_W), 0, 64, 0, P(G1B_B),
         t2, 64L*64, 64, 64, 256, 64, 1, 1.f);
    k_rowmax<<<cdiv(BN*64,256),256,0,stream>>>(t2, 64L*64, 64, 64, emb, 144, 0);
    tot = (long)BN*128*515;
    k_concat2<<<cdiv(tot,256),256,0,stream>>>(X, SX, 3, f4, 128L*512, 512,
                                              ce2, 128, tot);
    gemm(stream, BN, ce2, 128L*515, 515, P(G2A_W), 0, 128, 0, P(G2A_B),
         t3, 128L*128, 128, 128, 515, 128, 1, 1.f);
    gemm(stream, BN, t3, 128L*128, 128, P(G2B_W), 0, 64, 0, P(G2B_B),
         t4, 128L*64, 64, 128, 128, 64, 1, 1.f);
    k_rowmax<<<cdiv(BN*64,256),256,0,stream>>>(t4, 128L*64, 128, 64, emb, 144, 64);
    k_copyvec<<<cdiv(BN*CAT,64),64,0,stream>>>(L, CAT, emb, 144, 128, BN*CAT);
  }

  // ---- decoder: FP interpolation + diff-attention up-convs ----
  fp_mod(stream, X, 128, 64, f4, 128L*512, 512, f5, 64L*1024, 1024,
         P(FP4_W), P(FP4_B), 256, fp4o, 128L*256, scr);
  diff_conv(stream, fp4o, 128L*256, 128, X, 128, 256, 512,
            P(UP5_WQ), P(UP5_WK), P(UP5_OW), P(UP5_OB), 0.80f, up5o, 128L*512, scr);
  fp_mod(stream, X, 256, 128, f3, 256L*256, 256, up5o, 128L*512, 512,
         P(FP3_W), P(FP3_B), 128, fp3o, 256L*128, scr);
  diff_conv(stream, fp3o, 256L*128, 256, X, 256, 128, 256,
            P(UP4_WQ), P(UP4_WK), P(UP4_OW), P(UP4_OB), 0.40f, up4o, 256L*256, scr);
  fp_mod(stream, X, 512, 256, f2, 512L*128, 128, up4o, 256L*256, 256,
         P(FP2_W), P(FP2_B), 64, fp2o, 512L*64, scr);
  diff_conv(stream, fp2o, 512L*64, 512, X, 512, 64, 128,
            P(UP3_WQ), P(UP3_WK), P(UP3_OW), P(UP3_OB), 0.20f, up3o, 512L*128, scr);
  fp_mod(stream, X, NPTS, 512, f1, (long)NPTS*64, 64, up3o, 512L*128, 128,
         P(FP1_W), P(FP1_B), 32, fp1o, (long)NPTS*32, scr);
  diff_conv(stream, fp1o, (long)NPTS*32, NPTS, X, NPTS, 32, 64,
            P(UP2_WQ), P(UP2_WK), P(UP2_OW), P(UP2_OB), 0.05f, up2o,
            (long)NPTS*64, scr);

  // ---- head: concat + up1 + SE gating + classifier + transpose ----
  long tot = (long)BN*NPTS*211;
  k_concat3<<<cdiv(tot,256),256,0,stream>>>(X, SX, 3, up2o, (long)NPTS*64, 64,
                                            emb, 144, 144, featc, NPTS, tot);
  gemm(stream, BN, featc, (long)NPTS*211, 211, P(UP1_W), 0, 256, 0, P(UP1_B),
       featg, (long)NPTS*256, 256, NPTS, 211, 256, 1, 1.f);
  k_mean_rows<<<cdiv(BN*256,256),256,0,stream>>>(featg, NPTS, 256, s0);
  gemm(stream, 1, s0, 0, 256, P(SE1), 0, 32, 0, nullptr,
       s1g, 0, 32, BN, 256, 32, 1, 1.f);                        // gelu
  gemm(stream, 1, s1g, 0, 32, P(SE2), 0, 256, 0, nullptr,
       s2s, 0, 256, BN, 32, 256, 3, 1.f);                       // sigmoid
  long ts = (long)BN*NPTS*256;
  k_rowscale<<<cdiv(ts,256),256,0,stream>>>(featg, s2s, feats, NPTS, 256, ts);
  gemm(stream, BN, feats, (long)NPTS*256, 256, P(LAST), 0, NCLS, 0, nullptr,
       lgt, (long)NPTS*NCLS, NCLS, NPTS, 256, NCLS, 0, 1.f);
  k_transpose_out<<<cdiv((long)BN*NPTS*NCLS,256),256,0,stream>>>(
      lgt, (float*)d_out, NPTS, NCLS);
#undef P
}
